// MultiHeadSlotAttention_4621384810958
// MI455X (gfx1250) — matrix operand order repaired and checked by reading, not compiled, not run on hardware
//
#include <hip/hip_runtime.h>

// MultiHeadSlotAttention for MI455X (gfx1250).
// bf16 WMMA (f32 accumulate), fused attention iteration in 320KB WGP LDS,
// async global->LDS staging, prefetch hints.
// B=64 N=4096 D=256 S=8 H=8 DH=32 HID=256 ITERS=3

#define B_    64
#define N_    4096
#define D_    256
#define S_    8
#define H_    8
#define DH_   32
#define ITERS_ 3

typedef __bf16 bf16;
typedef __attribute__((ext_vector_type(16))) __bf16 v16bf;
typedef __attribute__((ext_vector_type(8)))  float  v8f;

// ---------------------------------------------------------------- WMMA core
__device__ __forceinline__ v8f wmma_bf16f32(v16bf a, v16bf b, v8f c) {
  // (neg_a, A, neg_b, B, c_mod, C, reuse_a, reuse_b)
  return __builtin_amdgcn_wmma_f32_16x16x32_bf16(false, a, false, b, (short)0, c,
                                                 false, false);
}

// A-operand fragment (16x32 bf16, row-major, leading dim ld halves).
// lane<16: row r, K {0..7,16..23}; lane>=16: row r, K {8..15,24..31}.
__device__ __forceinline__ v16bf frag_a(const bf16* p, int ld, int lane) {
  int r = lane & 15, hi = (lane >> 4) & 1;
  const bf16* q = p + (size_t)r * ld + hi * 8;
  v16bf f;
#pragma unroll
  for (int i = 0; i < 8; ++i) f[i] = q[i];
#pragma unroll
  for (int i = 0; i < 8; ++i) f[8 + i] = q[16 + i];
  return f;
}

// Same but source has only 8 valid rows; rows 8..15 clamp (their D rows are
// discarded anyway - D row m depends only on A row m).
__device__ __forceinline__ v16bf frag_a8(const bf16* p, int ld, int lane) {
  int r = lane & 7, hi = (lane >> 4) & 1;
  const bf16* q = p + (size_t)r * ld + hi * 8;
  v16bf f;
#pragma unroll
  for (int i = 0; i < 8; ++i) f[i] = q[i];
#pragma unroll
  for (int i = 0; i < 8; ++i) f[8 + i] = q[16 + i];
  return f;
}

// B-operand fragment (Bt row-major [16 cols][K]).
// lane<16: column r, K {0..7,16..23}; lane>=16: column r, K {8..15,24..31}.
__device__ __forceinline__ v16bf frag_b(const bf16* p, int ld, int lane) {
  int r = lane & 15, hi = (lane >> 4) & 1;
  const bf16* q = p + (size_t)r * ld + hi * 8;
  v16bf f;
#pragma unroll
  for (int i = 0; i < 8; ++i) f[i] = q[i];
#pragma unroll
  for (int i = 0; i < 8; ++i) f[8 + i] = q[16 + i];
  return f;
}

// ------------------------------------------------- async global->LDS helpers
__device__ __forceinline__ void async_load_b128(unsigned int lds_addr,
                                                const void* gaddr) {
  asm volatile("global_load_async_to_lds_b128 %0, %1, off"
               :: "v"(lds_addr), "v"(gaddr) : "memory");
}
__device__ __forceinline__ void wait_asynccnt0() {
  asm volatile("s_wait_asynccnt 0x0" ::: "memory");
}

// ---------------------------------------------------------------- utilities
__global__ void zero_f32_kernel(float* p, int n) {
  int i = blockIdx.x * 256 + threadIdx.x;
  if (i < n) p[i] = 0.f;
}

__global__ void cast_bf16_kernel(const float* x, bf16* y, int n) {
  int i = blockIdx.x * 256 + threadIdx.x;
  if (i < n) y[i] = (bf16)x[i];
}

__global__ void copy_f32_kernel(const float* x, float* y, int n) {
  int i = blockIdx.x * 256 + threadIdx.x;
  if (i < n) y[i] = x[i];
}

// dst[c][r] = (bf16) src[r][c]   (src is [R][C] f32)
__global__ void transpose_bf16_kernel(const float* src, bf16* dst, int R, int C) {
  int idx = blockIdx.x * 256 + threadIdx.x;
  if (idx >= R * C) return;
  int r = idx / C, c = idx % C;
  dst[(size_t)c * R + r] = (bf16)src[idx];
}

// slots = mean + exp(logvar) * noise   [B,S,D] flat = 131072
__global__ void init_slots_kernel(const float* noise, const float* mean,
                                  const float* logv, float* slots) {
  int idx = blockIdx.x * 256 + threadIdx.x;
  int d = idx & 255;
  slots[idx] = mean[d] + __expf(logv[d]) * noise[idx];
}

// LayerNorm over rows of 256, output bf16. One wave per row.
__global__ void ln_rows_kernel(const float* x, const float* w, const float* b,
                               bf16* y, int R) {
  int wv = threadIdx.x >> 5, lane = threadIdx.x & 31;
  int row = blockIdx.x * 8 + wv;
  if (row >= R) return;
  const float* xp = x + (size_t)row * 256;
  float v[8];
  float s = 0.f;
#pragma unroll
  for (int i = 0; i < 8; ++i) { v[i] = xp[lane + 32 * i]; s += v[i]; }
#pragma unroll
  for (int o = 16; o > 0; o >>= 1) s += __shfl_xor(s, o, 32);
  float mean = s * (1.f / 256.f);
  float sq = 0.f;
#pragma unroll
  for (int i = 0; i < 8; ++i) { float d = v[i] - mean; sq += d * d; }
#pragma unroll
  for (int o = 16; o > 0; o >>= 1) sq += __shfl_xor(sq, o, 32);
  float rs = rsqrtf(sq * (1.f / 256.f) + 1e-5f);
#pragma unroll
  for (int i = 0; i < 8; ++i) {
    int c = lane + 32 * i;
    y[(size_t)row * 256 + c] = (bf16)((v[i] - mean) * rs * w[c] + b[c]);
  }
}

// -------------------------------------------------- fused LN + K/V projection
// Each block: 16 token rows of one batch. Raw tile staged via async->LDS,
// LN into LDS (bf16), then WMMA vs combined [wk|wv]^T.
// K rows -> klin[b][n][256], V -> vt[b][h][dh][n].
__global__ void lnproj_kernel(const float* inputs, const float* lnw, const float* lnb,
                              const bf16* wkvT, const float* bk, const float* bv,
                              bf16* klin, bf16* vt) {
  __shared__ float xraw[16 * 256];   // 16KB raw f32 tile (async staged)
  __shared__ bf16  xs[16 * 256];     // 8KB LN'd bf16 tile
  int tid = threadIdx.x;
  int w = tid >> 5, lane = tid & 31;
  int b = blockIdx.x >> 8;           // N/16 = 256 tiles per batch
  int tile = blockIdx.x & 255;
  int n0 = tile * 16;

  // async copy of the 16x256 f32 tile into LDS (ASYNCcnt-tracked)
  {
    const char* gsrc = (const char*)(inputs + ((size_t)(b * N_ + n0)) * 256);
    unsigned int lbase = (unsigned int)(size_t)(&xraw[0]);
#pragma unroll
    for (int rnd = 0; rnd < 4; ++rnd) {
      int off = (rnd * 256 + tid) * 16;
      async_load_b128(lbase + off, gsrc + off);
    }
    wait_asynccnt0();
  }
  __syncthreads();

#pragma unroll
  for (int rr = 0; rr < 2; ++rr) {
    int row = w * 2 + rr;
    float v[8];
    float s = 0.f;
#pragma unroll
    for (int i = 0; i < 8; ++i) { v[i] = xraw[row * 256 + lane + 32 * i]; s += v[i]; }
#pragma unroll
    for (int o = 16; o > 0; o >>= 1) s += __shfl_xor(s, o, 32);
    float mean = s * (1.f / 256.f);
    float sq = 0.f;
#pragma unroll
    for (int i = 0; i < 8; ++i) { float d = v[i] - mean; sq += d * d; }
#pragma unroll
    for (int o = 16; o > 0; o >>= 1) sq += __shfl_xor(sq, o, 32);
    float rs = rsqrtf(sq * (1.f / 256.f) + 1e-5f);
#pragma unroll
    for (int i = 0; i < 8; ++i) {
      int c = lane + 32 * i;
      xs[row * 256 + c] = (bf16)((v[i] - mean) * rs * lnw[c] + lnb[c]);
    }
  }
  __syncthreads();
  // 32 col tiles (256 k-outputs + 256 v-outputs), 4 per wave, K=256 -> 8 WMMA
#pragma unroll
  for (int t = 0; t < 4; ++t) {
    int ct = w + t * 8;
    int o0 = ct * 16;
    v8f c = {};
#pragma unroll
    for (int k0 = 0; k0 < 256; k0 += 32)
      c = wmma_bf16f32(frag_a(xs + k0, 256, lane),
                       frag_b(wkvT + (size_t)o0 * 256 + k0, 256, lane), c);
    int r = lane & 15, hi = (lane >> 4) & 1;
    int o = o0 + r;
    bool isv = o >= 256;
    int oo = isv ? o - 256 : o;
    float bias = isv ? bv[oo] : bk[oo];
#pragma unroll
    for (int j = 0; j < 8; ++j) {
      int n = n0 + hi * 8 + j;
      float val = c[j] + bias;
      if (!isv) {
        klin[((size_t)(b * N_ + n)) * 256 + oo] = (bf16)val;
      } else {
        int h = oo >> 5, dh = oo & 31;
        vt[(((size_t)(b * 8 + h)) * 32 + dh) * N_ + n] = (bf16)val;
      }
    }
  }
}

// ------------------------------------------------------- generic NT WMMA GEMM
// C[M,Nc] = A[M,K](bf16) * Bt[Nc,K](bf16)^T + bias; optional relu / residual.
// storeMode 0: row-major f32 and/or bf16; storeMode 2: q layout [b][h][8][32].
__global__ void gemm_nt_kernel(const bf16* A, int lda, const bf16* Bt, int ldb,
                               const float* bias, float* Cf, bf16* Cb, int ldc,
                               const float* resid, int relu, int storeMode,
                               int M, int Nc, int K) {
  int lane = threadIdx.x & 31;
  int wave = blockIdx.x * (blockDim.x >> 5) + (threadIdx.x >> 5);
  int ntn = Nc >> 4;
  int tiles = (M >> 4) * ntn;
  if (wave >= tiles) return;                 // uniform per-wave
  int mt = wave / ntn, nt = wave % ntn;
  const bf16* Ap = A + (size_t)mt * 16 * lda;
  const bf16* Bp = Bt + (size_t)nt * 16 * ldb;
  v8f c = {};
  for (int k0 = 0; k0 < K; k0 += 32)
    c = wmma_bf16f32(frag_a(Ap + k0, lda, lane), frag_b(Bp + k0, ldb, lane), c);
  int r = lane & 15, hi = (lane >> 4) & 1;
  int col = nt * 16 + r;
  float bs = bias ? bias[col] : 0.f;
#pragma unroll
  for (int j = 0; j < 8; ++j) {
    int row = mt * 16 + hi * 8 + j;
    float val = c[j] + bs;
    if (relu) val = fmaxf(val, 0.f);
    if (resid) val += resid[(size_t)row * ldc + col];
    if (storeMode == 0) {
      if (Cf) Cf[(size_t)row * ldc + col] = val;
      if (Cb) Cb[(size_t)row * ldc + col] = (bf16)val;
    } else {  // q: row=(b,s) col=(h,dh) -> qpad[b][h][8][32]
      int bb = row >> 3, ss = row & 7;
      int h = col >> 5, dh = col & 31;
      Cb[(((size_t)(bb * 8 + h)) * 8 + ss) * 32 + dh] = (bf16)val;
    }
  }
}

// ------------------------------------------- fused attention iteration kernel
// One block per (b,h). Phase 1: dots(WMMA K=32) + slot-softmax in registers,
// unnormalized p -> LDS bf16 (+ f32 spill on last iter). Phase 2: block-wide
// Sum_n p. Phase 3: upd = p @ v via WMMA over K=N, rows scaled by 1/sum
// (renorm is linear so it commutes with the GEMM).
__global__ void attn_iter_kernel(const bf16* qpad, const bf16* klin, const bf16* vt,
                                 float* updi, float* sums_g, float* p_out,
                                 int want_attn) {
  __shared__ bf16 pls[8 * 4096];     // 64KB unnormalized attn (+EPS), bf16
  __shared__ float redw[8 * 8];
  __shared__ float sums_l[8];
  int w = threadIdx.x >> 5, lane = threadIdx.x & 31;
  int bh = blockIdx.x;
  int b = bh >> 3, h = bh & 7;
  int r = lane & 15, hi = (lane >> 4) & 1;

  // q A-fragment is invariant for the whole block
  v16bf aq = frag_a8(qpad + (size_t)bh * 8 * 32, 32, lane);

  float ps[8];
#pragma unroll
  for (int j = 0; j < 8; ++j) ps[j] = 0.f;

  // ---- phase 1: dots + softmax over slots (register resident)
  for (int t = 0; t < 32; ++t) {
    int n0 = (w * 32 + t) * 16;
    const bf16* Bp = klin + ((size_t)(b * N_ + n0)) * 256 + h * 32;
    v8f c = {};
    c = wmma_bf16f32(aq, frag_b(Bp, 256, lane), c);
    if (hi == 0) {                    // lane r holds all 8 slot values of col n
      int n = n0 + r;
      float d[8], m = -1e30f;
#pragma unroll
      for (int j = 0; j < 8; ++j) { d[j] = c[j] * 0.0625f; m = fmaxf(m, d[j]); }
      float sum = 0.f;
#pragma unroll
      for (int j = 0; j < 8; ++j) { d[j] = __expf(d[j] - m); sum += d[j]; }
      float inv = 1.f / sum;
#pragma unroll
      for (int j = 0; j < 8; ++j) {
        float p = d[j] * inv + 1e-8f;  // softmax + EPS (unnormalized over N)
        ps[j] += p;
        pls[j * 4096 + n] = (bf16)p;
        if (want_attn) p_out[((size_t)(bh * 8 + j)) * N_ + n] = p;
      }
    }
  }
  // ---- phase 2: block reduction of Sum_n p per slot
#pragma unroll
  for (int j = 0; j < 8; ++j) {
    float v = ps[j];
#pragma unroll
    for (int o = 16; o > 0; o >>= 1) v += __shfl_xor(v, o, 32);
    if (lane == 0) redw[w * 8 + j] = v;
  }
  __syncthreads();
  if (w == 0 && lane < 8) {
    float s = 0.f;
#pragma unroll
    for (int k = 0; k < 8; ++k) s += redw[k * 8 + lane];
    sums_l[lane] = s;
    sums_g[bh * 8 + lane] = s;
  }
  __syncthreads();

  // ---- phase 3: upd = p @ v, K = N split over 4 chunk-waves x 2 dh-tiles
  int ct = w & 1, ch = w >> 1;
  const bf16* Bp = vt + ((size_t)bh * 32 + ct * 16) * N_;
  v8f c = {};
  for (int s = 0; s < 32; ++s) {
    int k0 = ch * 1024 + s * 32;
    __builtin_prefetch(Bp + k0 + 256, 0, 0);      // stream-ahead hint
    c = wmma_bf16f32(frag_a8(pls + k0, 4096, lane), frag_b(Bp + k0, N_, lane), c);
  }
  if (hi == 0) {
    int col = h * 32 + ct * 16 + r;
#pragma unroll
    for (int j = 0; j < 8; ++j) {
      float scale = 1.f / fmaxf(sums_l[j], 1e-12f);   // linear renorm
      atomicAdd(&updi[(size_t)(b * 8 + j) * 256 + col], c[j] * scale);
    }
  }
}

// final attn output: attn = p / sum  (only last iteration's p is kept)
__global__ void renorm_final_kernel(const float* p, const float* sums_g,
                                    float* attn_out) {
  int bh = blockIdx.x >> 4;
  int n = ((blockIdx.x & 15) << 8) + threadIdx.x;
  size_t base = (size_t)bh * 8 * N_ + n;
#pragma unroll
  for (int s = 0; s < 8; ++s) {
    float denom = fmaxf(sums_g[bh * 8 + s], 1e-12f);
    attn_out[base + (size_t)s * N_] = p[base + (size_t)s * N_] / denom;
  }
}

// GRU combine (torch gate order r,z,n)
__global__ void gru_kernel(const float* xg, const float* hg,
                           const float* slots_prev, float* out) {
  int idx = blockIdx.x * 256 + threadIdx.x;    // 512*256
  int row = idx >> 8, col = idx & 255;
  size_t base = (size_t)row * 768 + col;
  float r = 1.f / (1.f + __expf(-(xg[base] + hg[base])));
  float z = 1.f / (1.f + __expf(-(xg[base + 256] + hg[base + 256])));
  float nn = tanhf(xg[base + 512] + r * hg[base + 512]);
  float sp = slots_prev[idx];
  out[idx] = (1.f - z) * nn + z * sp;
}

// ------------------------------------------------------------------- launch
extern "C" void kernel_launch(void* const* d_in, const int* in_sizes, int n_in,
                              void* d_out, int out_size, void* d_ws, size_t ws_size,
                              hipStream_t stream) {
  const float* inputs = (const float*)d_in[0];
  const float* noise  = (const float*)d_in[1];
  const float* smean  = (const float*)d_in[2];
  const float* slogv  = (const float*)d_in[3];
  const float* wq = (const float*)d_in[4];   const float* bq = (const float*)d_in[5];
  const float* wk = (const float*)d_in[6];   const float* bk = (const float*)d_in[7];
  const float* wv = (const float*)d_in[8];   const float* bv = (const float*)d_in[9];
  const float* wo = (const float*)d_in[10];  const float* bo = (const float*)d_in[11];
  const float* w_ih = (const float*)d_in[12]; const float* b_ih = (const float*)d_in[13];
  const float* w_hh = (const float*)d_in[14]; const float* b_hh = (const float*)d_in[15];
  const float* w1 = (const float*)d_in[16];  const float* b1 = (const float*)d_in[17];
  const float* w2 = (const float*)d_in[18];  const float* b2 = (const float*)d_in[19];
  const float* ln_in_w = (const float*)d_in[20]; const float* ln_in_b = (const float*)d_in[21];
  const float* ln_s_w  = (const float*)d_in[22]; const float* ln_s_b  = (const float*)d_in[23];
  const float* ln_ff_w = (const float*)d_in[24]; const float* ln_ff_b = (const float*)d_in[25];

  float* out = (float*)d_out;
  float* attn_out = out + B_ * S_ * D_;       // slots first, then attn [B,H,S,N]

  char* ws = (char*)d_ws;
  size_t off = 0;
  auto alloc = [&](size_t bytes) -> char* {
    char* p = ws + off;
    off = (off + bytes + 255) & ~(size_t)255;
    return p;
  };
  bf16*  klin   = (bf16*)alloc((size_t)B_ * N_ * 256 * 2);       // [b][n][h*32+dh]
  bf16*  vt     = (bf16*)alloc((size_t)B_ * N_ * 256 * 2);       // [b][h][dh][n]
  float* pbuf   = (float*)alloc((size_t)B_ * H_ * S_ * N_ * 4);  // last-iter p
  bf16*  qpad   = (bf16*)alloc((size_t)B_ * H_ * 8 * 32 * 2);
  float* sums_g = (float*)alloc(4096 * 4);
  float* slots  = (float*)alloc(131072 * 4);
  float* slotsA = (float*)alloc(131072 * 4);
  bf16*  slots_b = (bf16*)alloc(131072 * 2);
  bf16*  sln_b   = (bf16*)alloc(131072 * 2);
  float* updi    = (float*)alloc(131072 * 4);
  bf16*  updi_b  = (bf16*)alloc(131072 * 2);
  bf16*  u2_b    = (bf16*)alloc(131072 * 2);
  float* xg      = (float*)alloc(393216 * 4);
  float* hg      = (float*)alloc(393216 * 4);
  bf16*  ffln_b  = (bf16*)alloc(131072 * 2);
  bf16*  h1_b    = (bf16*)alloc(131072 * 2);
  bf16*  wkvT = (bf16*)alloc(512 * 256 * 2);
  bf16*  wqT  = (bf16*)alloc(65536 * 2);
  bf16*  woT  = (bf16*)alloc(65536 * 2);
  bf16*  wihT = (bf16*)alloc(196608 * 2);
  bf16*  whhT = (bf16*)alloc(196608 * 2);
  bf16*  w1T  = (bf16*)alloc(65536 * 2);
  bf16*  w2T  = (bf16*)alloc(65536 * 2);

  dim3 blk(256);

  // weight transposes (to bf16 [out][in])
  transpose_bf16_kernel<<<256, blk, 0, stream>>>(wk, wkvT, 256, 256);
  transpose_bf16_kernel<<<256, blk, 0, stream>>>(wv, wkvT + 256 * 256, 256, 256);
  transpose_bf16_kernel<<<256, blk, 0, stream>>>(wq, wqT, 256, 256);
  transpose_bf16_kernel<<<256, blk, 0, stream>>>(wo, woT, 256, 256);
  transpose_bf16_kernel<<<768, blk, 0, stream>>>(w_ih, wihT, 256, 768);
  transpose_bf16_kernel<<<768, blk, 0, stream>>>(w_hh, whhT, 256, 768);
  transpose_bf16_kernel<<<256, blk, 0, stream>>>(w1, w1T, 256, 256);
  transpose_bf16_kernel<<<256, blk, 0, stream>>>(w2, w2T, 256, 256);

  init_slots_kernel<<<512, blk, 0, stream>>>(noise, smean, slogv, slots);

  // fused LN(inputs) + K/V projection (the big memory-bound GEMM)
  lnproj_kernel<<<B_ * (N_ / 16), blk, 0, stream>>>(inputs, ln_in_w, ln_in_b,
                                                    wkvT, bk, bv, klin, vt);

  for (int it = 0; it < ITERS_; ++it) {
    cast_bf16_kernel<<<512, blk, 0, stream>>>(slots, slots_b, 131072);
    ln_rows_kernel<<<64, blk, 0, stream>>>(slots, ln_s_w, ln_s_b, sln_b, 512);
    // q = LN(slots) @ wq + bq  -> [b][h][8][32] bf16
    gemm_nt_kernel<<<64, blk, 0, stream>>>(sln_b, 256, wqT, 256, bq,
                                           nullptr, qpad, 256, nullptr, 0, 2,
                                           512, 256, 256);
    zero_f32_kernel<<<512, blk, 0, stream>>>(updi, 131072);
    // fused dots + softmax + (linear) renorm + upd
    attn_iter_kernel<<<B_ * H_, blk, 0, stream>>>(qpad, klin, vt, updi, sums_g,
                                                  pbuf, it == ITERS_ - 1 ? 1 : 0);
    cast_bf16_kernel<<<512, blk, 0, stream>>>(updi, updi_b, 131072);
    // upd @ wo + bo
    gemm_nt_kernel<<<64, blk, 0, stream>>>(updi_b, 256, woT, 256, bo,
                                           nullptr, u2_b, 256, nullptr, 0, 0,
                                           512, 256, 256);
    // GRU gates
    gemm_nt_kernel<<<192, blk, 0, stream>>>(u2_b, 256, wihT, 256, b_ih,
                                            xg, nullptr, 768, nullptr, 0, 0,
                                            512, 768, 256);
    gemm_nt_kernel<<<192, blk, 0, stream>>>(slots_b, 256, whhT, 256, b_hh,
                                            hg, nullptr, 768, nullptr, 0, 0,
                                            512, 768, 256);
    gru_kernel<<<512, blk, 0, stream>>>(xg, hg, slots, slotsA);
    // FF block with residual
    ln_rows_kernel<<<64, blk, 0, stream>>>(slotsA, ln_ff_w, ln_ff_b, ffln_b, 512);
    gemm_nt_kernel<<<64, blk, 0, stream>>>(ffln_b, 256, w1T, 256, b1,
                                           nullptr, h1_b, 256, nullptr, 1, 0,
                                           512, 256, 256);
    gemm_nt_kernel<<<64, blk, 0, stream>>>(h1_b, 256, w2T, 256, b2,
                                           slots, nullptr, 256, slotsA, 0, 0,
                                           512, 256, 256);
  }
  renorm_final_kernel<<<B_ * H_ * (N_ / 256), blk, 0, stream>>>(pbuf, sums_g,
                                                                attn_out);
  copy_f32_kernel<<<512, blk, 0, stream>>>(slots, out, 131072);
  (void)in_sizes; (void)n_in; (void)out_size; (void)ws_size;
}
